// EncoderLayer_68779606278853
// MI455X (gfx1250) — compile-verified, matrix operand order checked by reading, not run on hardware
//
#include <hip/hip_runtime.h>

#define DEVINL __device__ __forceinline__

typedef __attribute__((ext_vector_type(16))) __bf16       bf16x16;
typedef __attribute__((ext_vector_type(8)))  float        f32x8;
typedef __attribute__((ext_vector_type(4)))  unsigned int u32x4;

namespace {
constexpr int   NTOK = 256;          // sequence length N
constexpr int   EDIM = 256;          // embedding E
constexpr int   NH   = 8;            // heads
constexpr int   DKH  = 32;           // head dim
constexpr float SCALING = 0.17677669529663687f;  // DK^-0.5
constexpr float EPS = 1e-5f;

// LDS pitches in bf16 elements. Chosen so the row stride in dwords is
// congruent to 4 (mod 64 banks) -> ds_load_b128 A/B fragments hit all 64 banks.
constexpr int XP  = 264;   // X tile / score-strip / V^T pitch (528 B rows)
constexpr int KP  = 40;    // Q/K strip pitch (80 B rows)
constexpr int PSP = 264;
constexpr int VTP = 264;
constexpr int TSP = 1032;  // FFN hidden strip pitch (2064 B rows)

// kernel-1 LDS element offsets (ushort units), all 16B aligned
constexpr int LS_XS  = 0;                       // [256][264]
constexpr int LS_KS  = LS_XS  + 256 * XP;       // [256][40]
constexpr int LS_VT  = LS_KS  + 256 * KP;       // [32][264]  (V transposed)
constexpr int LS_QW  = LS_VT  + 32  * VTP;      // 8 waves x [16][40]
constexpr int LS_PS  = LS_QW  + 8 * 16 * KP;    // 8 waves x [16][264]
constexpr int LS1_END = LS_PS + 8 * 16 * PSP;   // 125184 elems = 250368 B
constexpr int SMEM1 = LS1_END * 2;

constexpr int LS_X1S = 0;                       // [64][264]
constexpr int LS_TS  = LS_X1S + 64 * XP;        // [64][1032]
constexpr int SMEM2 = (LS_TS + 64 * TSP) * 2;   // 165888 B

// workspace layout (bytes) — everything bf16, ~69 MB total
constexpr size_t OFF_WQT = 0;
constexpr size_t OFF_WKT = OFF_WQT + (size_t)256 * 256 * 2;
constexpr size_t OFF_WVT = OFF_WKT + (size_t)256 * 256 * 2;
constexpr size_t OFF_WOT = OFF_WVT + (size_t)256 * 256 * 2;
constexpr size_t OFF_W1T = OFF_WOT + (size_t)256 * 256 * 2;
constexpr size_t OFF_W2T = OFF_W1T + (size_t)256 * 1024 * 2;
constexpr size_t OFF_XB  = OFF_W2T + (size_t)1024 * 256 * 2;  // bf16 x   [256][256][256]
constexpr size_t OFF_X1  = OFF_XB  + (size_t)256 * 256 * 256 * 2;  // bf16 x1
}  // namespace

DEVINL unsigned short f2bf(float f) {            // round-to-nearest-even f32->bf16
  unsigned int u = __float_as_uint(f);
  u += 0x7FFFu + ((u >> 16) & 1u);
  return (unsigned short)(u >> 16);
}
DEVINL float bf2f(unsigned short h) { return __uint_as_float(((unsigned int)h) << 16); }

// Async global->LDS copy of one 16B chunk per lane (CDNA5 path, ASYNCcnt).
// ldsoff is the LDS byte address (groupstaticsize + dynamic-segment offset).
DEVINL void async_g2l_b128(unsigned ldsoff, const void* gaddr) {
  asm volatile("global_load_async_to_lds_b128 %0, %1, off"
               :: "v"(ldsoff), "v"(gaddr) : "memory");
}
DEVINL void wait_async0() { asm volatile("s_wait_asynccnt 0" ::: "memory"); }

// Load a 16x32 bf16 A/B fragment from a row-major bf16 matrix (`pitch` elems).
// CDNA5 16-bit layout: lane<16 holds row (rowb+lane), K chunks {k0..k0+7, k0+16..k0+23};
// lane>=16 holds the other two 8-element chunks. Two ds/global b128 loads per lane.
DEVINL bf16x16 ldfrag(const unsigned short* __restrict__ buf, int pitch, int rowb, int k0) {
  const int lane = threadIdx.x & 31;
  const unsigned short* p = buf + (rowb + (lane & 15)) * pitch + k0 + ((lane >> 4) << 3);
  struct Pk { u32x4 lo, hi; } pk;
  pk.lo = *(const u32x4*)(p);
  pk.hi = *(const u32x4*)(p + 16);
  return __builtin_bit_cast(bf16x16, pk);
}

DEVINL f32x8 wmma_bf16(bf16x16 a, bf16x16 b, f32x8 c) {
  return __builtin_amdgcn_wmma_f32_16x16x32_bf16(false, a, false, b, (short)0, c, false, false);
}

// C-fragment row reductions: row M=r+8*(lane>=16) is spread over a 16-lane half,
// so xor-butterfly with masks 8..1 stays inside the half. wave32 only.
DEVINL void redmax8(float v[8]) {
#pragma unroll
  for (int m = 8; m; m >>= 1) {
#pragma unroll
    for (int r = 0; r < 8; ++r) v[r] = fmaxf(v[r], __shfl_xor(v[r], m, 32));
  }
}
DEVINL void redsum8(float v[8]) {
#pragma unroll
  for (int m = 8; m; m >>= 1) {
#pragma unroll
    for (int r = 0; r < 8; ++r) v[r] += __shfl_xor(v[r], m, 32);
  }
}

// ---------------- weight transpose + bf16 convert ----------------
__global__ void __launch_bounds__(256)
transpose_to_bf16(const float* __restrict__ src, unsigned short* __restrict__ dst,
                  int R, int C) {
  int idx = blockIdx.x * 256 + threadIdx.x;
  if (idx >= R * C) return;
  int r = idx / C, c = idx - r * C;
  dst[c * R + r] = f2bf(src[idx]);   // coalesced read, transposed bf16 write
}

// ---------------- x (f32) -> bf16, so kernels can async-DMA it into LDS ------
__global__ void __launch_bounds__(256)
convert_to_bf16(const float* __restrict__ src, unsigned short* __restrict__ dst) {
  int idx = (blockIdx.x * 256 + threadIdx.x) * 4;
  float4 v = *(const float4*)(src + idx);
  ushort4 o;
  o.x = f2bf(v.x); o.y = f2bf(v.y); o.z = f2bf(v.z); o.w = f2bf(v.w);
  *(ushort4*)(dst + idx) = o;
}

// ---------------- fused attention + out-proj + residual + LN1 ----------------
// grid.x = 512: seq = bx>>1 (256 sequences), query-half = bx&1 (128 rows each).
// 8 waves: each owns 16 query rows; K/V strips computed cooperatively per head.
__global__ void __launch_bounds__(256)
enc_attn_ln1(const unsigned short* __restrict__ xb, const float* __restrict__ mask,
             const float* __restrict__ bo, const float* __restrict__ g1,
             const float* __restrict__ beta1,
             const unsigned short* __restrict__ WqT, const unsigned short* __restrict__ WkT,
             const unsigned short* __restrict__ WvT, const unsigned short* __restrict__ WoT,
             unsigned short* __restrict__ x1) {
  extern __shared__ char smem[];
  unsigned short* sm  = (unsigned short*)smem;
  unsigned short* Xs  = sm + LS_XS;
  unsigned short* Ks  = sm + LS_KS;
  unsigned short* VTs = sm + LS_VT;

  const int tid = threadIdx.x;
  const int lane = tid & 31, wave = tid >> 5;
  const int half = lane >> 4, ln = lane & 15;
  const int seq = blockIdx.x >> 1;
  const int q0 = (blockIdx.x & 1) * 128 + wave * 16;   // this wave's query rows

  unsigned short* Qs = sm + LS_QW + wave * 16 * KP;
  unsigned short* Pw = sm + LS_PS + wave * 16 * PSP;

  // ---- stage X (bf16) into pitched LDS via async tensor-path DMA ----
  const unsigned short* xg = xb + (size_t)seq * (NTOK * EDIM);
  const unsigned lds0 = __builtin_amdgcn_groupstaticsize();  // dynamic LDS base
#pragma unroll 4
  for (int i = 0; i < 32; ++i) {
    int e = (i * 256 + tid) * 8;                 // 8 bf16 = 16 B per lane
    int r = e >> 8, c = e & 255;
    async_g2l_b128(lds0 + (unsigned)((LS_XS + r * XP + c) * 2), xg + e);
  }
  wait_async0();
  __syncthreads();

  const float* maskp = mask + (size_t)(seq & 15) * (NTOK * NTOK);
  f32x8 zf = {};
  f32x8 oacc[16];                        // attn@Wo strip accumulator (16 x 256)
#pragma unroll
  for (int nt = 0; nt < 16; ++nt) oacc[nt] = zf;

  for (int h = 0; h < NH; ++h) {
    const int hc = h * DKH;
    // ---- Q strip (own 16 rows x 32), pre-scaled by 1/sqrt(dk) ----
#pragma unroll
    for (int nt = 0; nt < 2; ++nt) {
      f32x8 acc = zf;
#pragma unroll
      for (int kk = 0; kk < 8; ++kk)
        acc = wmma_bf16(ldfrag(Xs, XP, q0, kk * 32),
                        ldfrag(WqT, 256, hc + nt * 16, kk * 32), acc);
#pragma unroll
      for (int r = 0; r < 8; ++r)
        Qs[(half * 8 + r) * KP + nt * 16 + ln] = f2bf(acc[r] * SCALING);
    }
    // ---- K rows / V^T columns for keys [wave*32, wave*32+32) ----
#pragma unroll
    for (int s = 0; s < 2; ++s) {
      const int kb = wave * 32 + s * 16;
#pragma unroll
      for (int nt = 0; nt < 2; ++nt) {
        f32x8 acc = zf;
#pragma unroll
        for (int kk = 0; kk < 8; ++kk)
          acc = wmma_bf16(ldfrag(Xs, XP, kb, kk * 32),
                          ldfrag(WkT, 256, hc + nt * 16, kk * 32), acc);
#pragma unroll
        for (int r = 0; r < 8; ++r)
          Ks[(kb + half * 8 + r) * KP + nt * 16 + ln] = f2bf(acc[r]);
      }
#pragma unroll
      for (int nt = 0; nt < 2; ++nt) {
        f32x8 acc = zf;
#pragma unroll
        for (int kk = 0; kk < 8; ++kk)
          acc = wmma_bf16(ldfrag(Xs, XP, kb, kk * 32),
                          ldfrag(WvT, 256, hc + nt * 16, kk * 32), acc);
#pragma unroll
        for (int r = 0; r < 8; ++r)  // store transposed: VTs[d][token]
          VTs[(nt * 16 + ln) * VTP + kb + half * 8 + r] = f2bf(acc[r]);
      }
    }
    __syncthreads();

    // ---- pass 1: S = QK^T + mask, track row max, spill bf16 strip ----
    float mx[8];
#pragma unroll
    for (int r = 0; r < 8; ++r) mx[r] = -3.0e38f;
    bf16x16 qa = ldfrag(Qs, KP, 0, 0);           // A reused across all 16 tiles
    for (int t = 0; t < 16; ++t) {
      f32x8 s = wmma_bf16(qa, ldfrag(Ks, KP, t * 16, 0), zf);   // single K=32 step
#pragma unroll
      for (int r = 0; r < 8; ++r) {
        float v = s[r] + maskp[(q0 + half * 8 + r) * NTOK + t * 16 + ln];
        mx[r] = fmaxf(mx[r], v);
        Pw[(half * 8 + r) * PSP + t * 16 + ln] = f2bf(v);
      }
    }
    redmax8(mx);
    // ---- pass 2: exponentiate in place, row sums ----
    float ls[8];
#pragma unroll
    for (int r = 0; r < 8; ++r) ls[r] = 0.f;
    for (int t = 0; t < 16; ++t) {
#pragma unroll
      for (int r = 0; r < 8; ++r) {
        unsigned short* pp = Pw + (half * 8 + r) * PSP + t * 16 + ln;
        float p = __expf(bf2f(*pp) - mx[r]);
        ls[r] += p;
        *pp = f2bf(p);
      }
    }
    redsum8(ls);
#pragma unroll
    for (int r = 0; r < 8; ++r) ls[r] = 1.0f / ls[r];

    // ---- context = P @ V (16 x 32) ----
    f32x8 ctx0 = zf, ctx1 = zf;
#pragma unroll
    for (int kk = 0; kk < 8; ++kk) {
      bf16x16 pa = ldfrag(Pw, PSP, 0, kk * 32);
      ctx0 = wmma_bf16(pa, ldfrag(VTs, VTP, 0, kk * 32), ctx0);
      ctx1 = wmma_bf16(pa, ldfrag(VTs, VTP, 16, kk * 32), ctx1);
    }
#pragma unroll
    for (int r = 0; r < 8; ++r) {                 // normalize + bounce via LDS
      Pw[(half * 8 + r) * PSP + ln]      = f2bf(ctx0[r] * ls[r]);
      Pw[(half * 8 + r) * PSP + 16 + ln] = f2bf(ctx1[r] * ls[r]);
    }
    // ---- fused out-projection: oacc += ctx @ Wo[h*32:(h+1)*32, :] ----
    bf16x16 ca = ldfrag(Pw, PSP, 0, 0);
#pragma unroll
    for (int nt = 0; nt < 16; ++nt)
      oacc[nt] = wmma_bf16(ca, ldfrag(WoT, 256, nt * 16, hc), oacc[nt]);
    __syncthreads();                              // before next head reuses Ks/VTs
  }

  // ---- residual + bias + LayerNorm1, write x1 (bf16) ----
  float sm8[8], sq8[8];
#pragma unroll
  for (int r = 0; r < 8; ++r) { sm8[r] = 0.f; sq8[r] = 0.f; }
#pragma unroll
  for (int nt = 0; nt < 16; ++nt) {
    const int col = nt * 16 + ln;
    const float bov = bo[col];
#pragma unroll
    for (int r = 0; r < 8; ++r) {
      float v = bf2f(Xs[(q0 + half * 8 + r) * XP + col]) + oacc[nt][r] + bov;
      oacc[nt][r] = v;
      sm8[r] += v; sq8[r] += v * v;
    }
  }
  redsum8(sm8); redsum8(sq8);
  float mu[8], rs[8];
#pragma unroll
  for (int r = 0; r < 8; ++r) {
    float m = sm8[r] * (1.0f / EDIM);
    mu[r] = m;
    rs[r] = rsqrtf(sq8[r] * (1.0f / EDIM) - m * m + EPS);
  }
  unsigned short* xo = x1 + (size_t)seq * (NTOK * EDIM);
#pragma unroll
  for (int nt = 0; nt < 16; ++nt) {
    const int col = nt * 16 + ln;
    const float gg = g1[col], bb = beta1[col];
#pragma unroll
    for (int r = 0; r < 8; ++r)
      xo[(q0 + half * 8 + r) * EDIM + col] =
          f2bf((oacc[nt][r] - mu[r]) * rs[r] * gg + bb);
  }
}

// ---------------- fused FFN + residual + LN2 ----------------
// grid.x = 1024: seq = bx>>2, 64-row block = bx&3. 4 waves x 16 rows.
// hidden strip relu(x1@W1+b1) lives only in LDS (never touches HBM).
__global__ void __launch_bounds__(128)
enc_ffn_ln2(const unsigned short* __restrict__ x1, const float* __restrict__ bf1,
            const float* __restrict__ bf2, const float* __restrict__ g2,
            const float* __restrict__ beta2,
            const unsigned short* __restrict__ W1T, const unsigned short* __restrict__ W2T,
            float* __restrict__ out) {
  extern __shared__ char smem[];
  unsigned short* x1s = (unsigned short*)smem + LS_X1S;   // [64][264]
  unsigned short* Ts  = (unsigned short*)smem + LS_TS;    // [64][1032]

  const int tid = threadIdx.x;
  const int lane = tid & 31, wave = tid >> 5;
  const int half = lane >> 4, ln = lane & 15;
  const int seq = blockIdx.x >> 2;
  const int wr = wave * 16;                                // strip-local rows
  const size_t gbase = (size_t)seq * (NTOK * EDIM) + (size_t)((blockIdx.x & 3) * 64) * EDIM;

  // ---- async-stage this wave's 16 rows of x1 (bf16); rows are wave-private ----
  const unsigned short* xg = x1 + gbase + (size_t)wr * EDIM;
  const unsigned lds0 = __builtin_amdgcn_groupstaticsize();
#pragma unroll 4
  for (int i = 0; i < 16; ++i) {
    int e = (i * 32 + lane) * 8;                 // 0..4088 within 16x256 block
    int r = e >> 8, c = e & 255;
    async_g2l_b128(lds0 + (unsigned)((LS_X1S + (wr + r) * XP + c) * 2), xg + e);
  }
  wait_async0();   // own rows only -> no barrier needed

  f32x8 zf = {};
  // ---- hidden = relu(x1 @ W1 + b1): 16 x 1024 strip into LDS ----
  for (int nt = 0; nt < 64; ++nt) {
    f32x8 acc = zf;
#pragma unroll
    for (int kk = 0; kk < 8; ++kk)
      acc = wmma_bf16(ldfrag(x1s, XP, wr, kk * 32),
                      ldfrag(W1T, 256, nt * 16, kk * 32), acc);
    const int col = nt * 16 + ln;
    const float b = bf1[col];
#pragma unroll
    for (int r = 0; r < 8; ++r)
      Ts[(wr + half * 8 + r) * TSP + col] = f2bf(fmaxf(acc[r] + b, 0.f));
  }

  // ---- y = hidden @ W2 (A-fragment reused across 16 output tiles) ----
  f32x8 oacc[16];
#pragma unroll
  for (int nt = 0; nt < 16; ++nt) oacc[nt] = zf;
  for (int kk = 0; kk < 32; ++kk) {
    bf16x16 ta = ldfrag(Ts, TSP, wr, kk * 32);
#pragma unroll
    for (int nt = 0; nt < 16; ++nt)
      oacc[nt] = wmma_bf16(ta, ldfrag(W2T, 1024, nt * 16, kk * 32), oacc[nt]);
  }

  // ---- residual + bias + LayerNorm2 -> out (f32) ----
  float sm8[8], sq8[8];
#pragma unroll
  for (int r = 0; r < 8; ++r) { sm8[r] = 0.f; sq8[r] = 0.f; }
#pragma unroll
  for (int nt = 0; nt < 16; ++nt) {
    const int col = nt * 16 + ln;
    const float bb = bf2[col];
#pragma unroll
    for (int r = 0; r < 8; ++r) {
      float v = bf2f(x1s[(wr + half * 8 + r) * XP + col]) + oacc[nt][r] + bb;
      oacc[nt][r] = v;
      sm8[r] += v; sq8[r] += v * v;
    }
  }
  redsum8(sm8); redsum8(sq8);
  float mu[8], rs[8];
#pragma unroll
  for (int r = 0; r < 8; ++r) {
    float m = sm8[r] * (1.0f / EDIM);
    mu[r] = m;
    rs[r] = rsqrtf(sq8[r] * (1.0f / EDIM) - m * m + EPS);
  }
  float* og = out + gbase;
#pragma unroll
  for (int nt = 0; nt < 16; ++nt) {
    const int col = nt * 16 + ln;
    const float gg = g2[col], bb = beta2[col];
#pragma unroll
    for (int r = 0; r < 8; ++r)
      og[(wr + half * 8 + r) * EDIM + col] = (oacc[nt][r] - mu[r]) * rs[r] * gg + bb;
  }
}

extern "C" void kernel_launch(void* const* d_in, const int* in_sizes, int n_in,
                              void* d_out, int out_size, void* d_ws, size_t ws_size,
                              hipStream_t stream) {
  (void)in_sizes; (void)n_in; (void)out_size; (void)ws_size;
  const float* x    = (const float*)d_in[0];
  const float* mask = (const float*)d_in[1];
  const float* Wq   = (const float*)d_in[2];
  const float* Wk   = (const float*)d_in[3];
  const float* Wv   = (const float*)d_in[4];
  const float* Wo   = (const float*)d_in[5];
  const float* bo   = (const float*)d_in[6];
  const float* W1   = (const float*)d_in[7];
  const float* bf1  = (const float*)d_in[8];
  const float* W2   = (const float*)d_in[9];
  const float* bf2  = (const float*)d_in[10];
  const float* g1   = (const float*)d_in[11];
  const float* b1   = (const float*)d_in[12];
  const float* g2   = (const float*)d_in[13];
  const float* b2   = (const float*)d_in[14];
  float* out = (float*)d_out;

  char* ws = (char*)d_ws;
  unsigned short* WqT = (unsigned short*)(ws + OFF_WQT);
  unsigned short* WkT = (unsigned short*)(ws + OFF_WKT);
  unsigned short* WvT = (unsigned short*)(ws + OFF_WVT);
  unsigned short* WoT = (unsigned short*)(ws + OFF_WOT);
  unsigned short* W1T = (unsigned short*)(ws + OFF_W1T);
  unsigned short* W2T = (unsigned short*)(ws + OFF_W2T);
  unsigned short* xb  = (unsigned short*)(ws + OFF_XB);
  unsigned short* x1b = (unsigned short*)(ws + OFF_X1);

  hipFuncSetAttribute(reinterpret_cast<const void*>(enc_attn_ln1),
                      hipFuncAttributeMaxDynamicSharedMemorySize, SMEM1);
  hipFuncSetAttribute(reinterpret_cast<const void*>(enc_ffn_ln2),
                      hipFuncAttributeMaxDynamicSharedMemorySize, SMEM2);

  transpose_to_bf16<<<256, 256, 0, stream>>>(Wq, WqT, 256, 256);
  transpose_to_bf16<<<256, 256, 0, stream>>>(Wk, WkT, 256, 256);
  transpose_to_bf16<<<256, 256, 0, stream>>>(Wv, WvT, 256, 256);
  transpose_to_bf16<<<256, 256, 0, stream>>>(Wo, WoT, 256, 256);
  transpose_to_bf16<<<1024, 256, 0, stream>>>(W1, W1T, 256, 1024);
  transpose_to_bf16<<<1024, 256, 0, stream>>>(W2, W2T, 1024, 256);
  convert_to_bf16<<<16384, 256, 0, stream>>>(x, xb);   // 16.7M elems, float4/ushort4

  enc_attn_ln1<<<512, 256, SMEM1, stream>>>(xb, mask, bo, g1, b1,
                                            WqT, WkT, WvT, WoT, x1b);
  enc_ffn_ln2<<<1024, 128, SMEM2, stream>>>(x1b, bf1, bf2, g2, b2, W1T, W2T, out);
}
